// WindowAttention_44590350467506
// MI455X (gfx1250) — matrix operand order repaired and checked by reading, not compiled, not run on hardware
//
#include <hip/hip_runtime.h>

// ---------------- problem constants ----------------
constexpr int DIMC     = 192;
constexpr int HEADS    = 6;
constexpr int HEAD_DIM = 32;
constexpr int NTOK     = 64;               // 8x8 window
constexpr float QK_SCALE = 0.17677669529663687f; // 32^-0.5

// ---------------- WMMA fragment types ----------------
typedef __attribute__((ext_vector_type(16))) __bf16 v16bf;
typedef __attribute__((ext_vector_type(8)))  __bf16 v8bf;
typedef __attribute__((ext_vector_type(8)))  float  v8f;

union FragU { v16bf v; v8bf h[2]; };

// A operand: 16x32 bf16 tile, row-major source (row stride = `stride` elements).
// Per lane: M=L&15; 16 consecutive K elements starting at half*16.
__device__ __forceinline__ v16bf load_a_frag(const __bf16* base, int stride,
                                             int row0, int k0, int lane) {
  const int m = lane & 15, hf = lane >> 4;
  const __bf16* p = base + (row0 + m) * stride + k0 + hf * 16;
  FragU u;
  u.h[0] = *(const v8bf*)(p);
  u.h[1] = *(const v8bf*)(p + 8);
  return u.v;
}

// B operand: 32x16 bf16 tile where B[k][n] = src[n0+n][k0+k], src row-major.
// Contiguous order used here (same as load_a_frag): lanes 0-15 hold K=0..15 (N=lane), lanes 16-31 hold K=16..31.
// -> per lane: 16 consecutive K elements of one source row.
__device__ __forceinline__ v16bf load_b_frag(const __bf16* base, int stride,
                                             int n0, int k0, int lane) {
  const int n = lane & 15, hf = lane >> 4;
  const __bf16* p = base + (n0 + n) * stride + k0 + hf * 16;
  FragU u;
  u.h[0] = *(const v8bf*)(p);
  u.h[1] = *(const v8bf*)(p + 8);
  return u.v;
}

__device__ __forceinline__ v8f wmma_bf16(v16bf a, v16bf b, v8f c) {
  return __builtin_amdgcn_wmma_f32_16x16x32_bf16(false, a, false, b,
                                                 (short)0, c, false, false);
}

// ---------------- LDS layout (dynamic) ----------------
constexpr int XS = 200;  // x row stride (bf16 elems), 400 B (16B aligned)
constexpr int QS = 392;  // q|k rows stride (384 cols + pad)
constexpr int VS = 72;   // v^T row stride (64 tokens + pad)
constexpr int SS = 72;   // scores row stride (f32)
constexpr int PS = 72;   // probs row stride (bf16)
constexpr int OS = 200;  // attn-out row stride (bf16)

constexpr int OFF_X  = 0;                               // 64*200*2 = 25600
constexpr int OFF_QK = OFF_X  + NTOK * XS * 2;          // 64*392*2 = 50176
constexpr int OFF_VT = OFF_QK + NTOK * QS * 2;          // 192*72*2 = 27648
constexpr int OFF_S  = OFF_VT + DIMC * VS * 2;          // 64*72*4  = 18432
constexpr int OFF_P  = OFF_S  + NTOK * SS * 4;          // 64*72*2  =  9216
constexpr int OFF_O  = OFF_P  + NTOK * PS * 2;          // 64*200*2 = 25600
constexpr int SMEM_BYTES = OFF_O + NTOK * OS * 2;       // = 156672

// ---------------- workspace layout ----------------
constexpr size_t WS_WQ    = 0;                               // 576*192 bf16
constexpr size_t WS_WP    = WS_WQ + (size_t)3*DIMC*DIMC*2;   // 192*192 bf16
constexpr size_t WS_BIAS  = WS_WP + (size_t)DIMC*DIMC*2;     // 6*64*64 f32
// total = 221184 + 73728 + 98304 = 393216 bytes

// ============ kernel 0: weight convert + bias gather ============
__global__ void prep_kernel(const float* __restrict__ qkv_w,
                            const float* __restrict__ proj_w,
                            const float* __restrict__ bias_table,
                            const int*   __restrict__ rel_idx,
                            __bf16* __restrict__ wq, __bf16* __restrict__ wp,
                            float* __restrict__ bias6) {
  int i = blockIdx.x * 256 + threadIdx.x;
  if (i < 3 * DIMC * DIMC) wq[i] = (__bf16)qkv_w[i];
  if (i < DIMC * DIMC)     wp[i] = (__bf16)proj_w[i];
  if (i < HEADS * NTOK * NTOK) {
    int h  = i / (NTOK * NTOK);
    int nm = i % (NTOK * NTOK);
    bias6[i] = bias_table[rel_idx[nm] * HEADS + h];
  }
}

// ============ kernel 1: fused window attention ============
__global__ __launch_bounds__(128)
void win_attn_kernel(const float* __restrict__ x,
                     const float* __restrict__ qkv_b,
                     const float* __restrict__ proj_b,
                     const __bf16* __restrict__ wq,
                     const __bf16* __restrict__ wp,
                     const float* __restrict__ bias6,
                     float* __restrict__ out) {
  extern __shared__ char smem[];
  __bf16* xs   = (__bf16*)(smem + OFF_X);
  __bf16* qkvs = (__bf16*)(smem + OFF_QK);   // cols 0..191 = q*scale, 192..383 = k
  __bf16* vts  = (__bf16*)(smem + OFF_VT);   // v transposed: [d(0..191)][token]
  float*  Sb   = (float* )(smem + OFF_S);
  __bf16* Pb   = (__bf16*)(smem + OFF_P);
  __bf16* Os   = (__bf16*)(smem + OFF_O);

  const int tid  = threadIdx.x;
  const int wave = tid >> 5;
  const int lane = tid & 31;
  const int lhalf = lane >> 4;
  const int ln   = lane & 15;
  const int wnd  = blockIdx.x;
  const float* xg = x + (size_t)wnd * NTOK * DIMC;

  // ---- Phase 1: stage x -> bf16 LDS (float4 loads, rows stay intact: 192%4==0)
  for (int i = tid; i < NTOK * DIMC / 4; i += 128) {
    float4 f = ((const float4*)xg)[i];
    int e = i * 4;
    int n = e / DIMC, c = e % DIMC;
    __bf16* d = xs + n * XS + c;
    d[0] = (__bf16)f.x; d[1] = (__bf16)f.y;
    d[2] = (__bf16)f.z; d[3] = (__bf16)f.w;
  }
  __syncthreads();

  // ---- Phase 2: QKV = X @ Wqkv^T + b  (64x576, K=192)
  // region 0 = q-tiles (j 0..11), 1 = k-tiles (j 12..23), 2 = v-tiles (j 24..35).
  // Region is compile-time -> straight-line store paths (no exec-mask ladders).
#pragma unroll
  for (int region = 0; region < 3; ++region) {
    for (int jt = 0; jt < 3; ++jt) {
      int j = region * 12 + wave * 3 + jt;   // N-tile 0..35, region-uniform
      v16bf bfr[6];
#pragma unroll
      for (int k = 0; k < 6; ++k)
        bfr[k] = load_b_frag(wq, DIMC, j * 16, k * 32, lane);
      float bv = qkv_b[j * 16 + ln];
      int col = j * 16 + ln;
#pragma unroll
      for (int m = 0; m < 4; ++m) {
        v8f acc;
#pragma unroll
        for (int r = 0; r < 8; ++r) acc[r] = bv;
#pragma unroll
        for (int k = 0; k < 6; ++k) {
          v16bf a = load_a_frag(xs, XS, m * 16, k * 32, lane);
          acc = wmma_bf16(a, bfr[k], acc);
        }
        if (region == 0) {            // q: fold in softmax scale
#pragma unroll
          for (int r = 0; r < 8; ++r) {
            int row = m * 16 + lhalf * 8 + r;
            qkvs[row * QS + col] = (__bf16)(acc[r] * QK_SCALE);
          }
        } else if (region == 1) {     // k: row-major at col offset 192
#pragma unroll
          for (int r = 0; r < 8; ++r) {
            int row = m * 16 + lhalf * 8 + r;
            qkvs[row * QS + col] = (__bf16)acc[r];
          }
        } else {                      // v: transposed [d][token]
          int d = col - 2 * DIMC;
#pragma unroll
          for (int r = 0; r < 8; ++r) {
            int row = m * 16 + lhalf * 8 + r;
            vts[d * VS + row] = (__bf16)acc[r];
          }
        }
      }
    }
  }
  __syncthreads();

  // ---- Phase 3: per-head attention
  for (int h = 0; h < HEADS; ++h) {
    // 3a: S = (q*scale) @ k^T + bias   (K=32 -> single WMMA per tile)
    {
      int mi = wave;
      v16bf a = load_a_frag(qkvs + h * HEAD_DIM, QS, mi * 16, 0, lane);
#pragma unroll
      for (int ni = 0; ni < 4; ++ni) {
        v16bf b = load_b_frag(qkvs + DIMC + h * HEAD_DIM, QS, ni * 16, 0, lane);
        v8f acc;
#pragma unroll
        for (int r = 0; r < 8; ++r)
          acc[r] = bias6[(h * NTOK + mi * 16 + lhalf * 8 + r) * NTOK + ni * 16 + ln];
        acc = wmma_bf16(a, b, acc);
#pragma unroll
        for (int r = 0; r < 8; ++r)
          Sb[(mi * 16 + lhalf * 8 + r) * SS + ni * 16 + ln] = acc[r];
      }
    }
    __syncthreads();

    // 3b: row softmax in f32, write probs as bf16
    if (tid < NTOK) {
      float* row = Sb + tid * SS;
      float mx = -3.0e38f;
#pragma unroll 8
      for (int m = 0; m < NTOK; ++m) mx = fmaxf(mx, row[m]);
      float sum = 0.f;
#pragma unroll 8
      for (int m = 0; m < NTOK; ++m) { float e = __expf(row[m] - mx); row[m] = e; sum += e; }
      float inv = 1.f / sum;
      __bf16* pr = Pb + tid * PS;
#pragma unroll 8
      for (int m = 0; m < NTOK; ++m) pr[m] = (__bf16)(row[m] * inv);
    }
    __syncthreads();

    // 3c: O_h = P @ V_h   (K=64 -> 2 k-steps; B from transposed v, contiguous)
    {
      int mi = wave;
#pragma unroll
      for (int nj = 0; nj < 2; ++nj) {
        v8f acc;
#pragma unroll
        for (int r = 0; r < 8; ++r) acc[r] = 0.f;
#pragma unroll
        for (int ks = 0; ks < 2; ++ks) {
          v16bf a = load_a_frag(Pb, PS, mi * 16, ks * 32, lane);
          v16bf b = load_b_frag(vts + h * HEAD_DIM * VS, VS, nj * 16, ks * 32, lane);
          acc = wmma_bf16(a, b, acc);
        }
#pragma unroll
        for (int r = 0; r < 8; ++r)
          Os[(mi * 16 + lhalf * 8 + r) * OS + h * HEAD_DIM + nj * 16 + ln] = (__bf16)acc[r];
      }
    }
    __syncthreads();
  }

  // ---- Phase 4: Y = O @ Wproj^T + b  (64x192, K=192), f32 out to HBM
  for (int jj = 0; jj < 3; ++jj) {
    int j = wave * 3 + jj;               // N-tile 0..11
    v16bf bfr[6];
#pragma unroll
    for (int k = 0; k < 6; ++k)
      bfr[k] = load_b_frag(wp, DIMC, j * 16, k * 32, lane);
    float bv = proj_b[j * 16 + ln];
    int col = j * 16 + ln;
#pragma unroll
    for (int m = 0; m < 4; ++m) {
      v8f acc;
#pragma unroll
      for (int r = 0; r < 8; ++r) acc[r] = bv;
#pragma unroll
      for (int k = 0; k < 6; ++k) {
        v16bf a = load_a_frag(Os, OS, m * 16, k * 32, lane);
        acc = wmma_bf16(a, bfr[k], acc);
      }
#pragma unroll
      for (int r = 0; r < 8; ++r) {
        int row = m * 16 + lhalf * 8 + r;
        out[((size_t)wnd * NTOK + row) * DIMC + col] = acc[r];
      }
    }
  }
}

// ---------------- launcher ----------------
extern "C" void kernel_launch(void* const* d_in, const int* in_sizes, int n_in,
                              void* d_out, int out_size, void* d_ws, size_t ws_size,
                              hipStream_t stream) {
  const float* x          = (const float*)d_in[0];
  const float* qkv_w      = (const float*)d_in[1];
  const float* qkv_b      = (const float*)d_in[2];
  const float* proj_w     = (const float*)d_in[3];
  const float* proj_b     = (const float*)d_in[4];
  const float* bias_table = (const float*)d_in[5];
  const int*   rel_idx    = (const int*)d_in[6];
  float* out = (float*)d_out;

  char* ws = (char*)d_ws;
  __bf16* wq    = (__bf16*)(ws + WS_WQ);
  __bf16* wp    = (__bf16*)(ws + WS_WP);
  float*  bias6 = (float*)(ws + WS_BIAS);

  int nwin = in_sizes[0] / (NTOK * DIMC);

  // kernel 0: small prep (weights -> bf16, bias gather); re-run every call
  // (deterministic, inputs are never mutated)
  int prep_elems = 3 * DIMC * DIMC;  // largest range
  prep_kernel<<<(prep_elems + 255) / 256, 256, 0, stream>>>(
      qkv_w, proj_w, bias_table, rel_idx, wq, wp, bias6);

  // main fused attention kernel: one 4-wave block per window
  (void)hipFuncSetAttribute((const void*)win_attn_kernel,
                            hipFuncAttributeMaxDynamicSharedMemorySize,
                            SMEM_BYTES);
  win_attn_kernel<<<nwin, 128, SMEM_BYTES, stream>>>(
      x, qkv_b, proj_b, wq, wp, bias6, out);
}
